// Attention_63264868270755
// MI455X (gfx1250) — matrix operand order repaired and checked by reading, not compiled, not run on hardware
//
#include <hip/hip_runtime.h>

// ---------------- problem constants ----------------
constexpr int Bsz   = 2;
constexpr int N     = 2048;
constexpr int DIM   = 1024;
constexpr int H     = 16;
constexpr int DH    = 64;
constexpr int Mm    = 512;            // xl memory length
constexpr int INNER = 1024;
constexpr int KV    = Mm + 1 + N;     // 2561
constexpr int KVP   = 2624;           // padded KV rows (mult of 64) for in-bounds tail reads
constexpr int ROWS  = Bsz * N;        // 4096

// ---------------- WMMA types ----------------
typedef __attribute__((ext_vector_type(16))) __bf16        v16bf;
typedef __attribute__((ext_vector_type(8)))  float         v8f;
typedef __attribute__((ext_vector_type(4)))  unsigned int  u32x4;

union FragB16 { v16bf v; u32x4 u[2]; };

#define WMMA_BF16(a, b, c) \
  __builtin_amdgcn_wmma_f32_16x16x32_bf16(false, (a), false, (b), (short)0, (c), false, false)

__device__ __forceinline__ void wait_dscnt0() {
#if __has_builtin(__builtin_amdgcn_s_wait_dscnt)
  __builtin_amdgcn_s_wait_dscnt(0);
#else
  asm volatile("s_wait_dscnt 0" ::: "memory");
#endif
}

// width-16 xor reductions via ds_swizzle (stays inside 16-lane groups: matches C-layout halves)
__device__ __forceinline__ float redmax16(float x) {
  x = fmaxf(x, __int_as_float(__builtin_amdgcn_ds_swizzle(__float_as_int(x), 0x041F)));
  x = fmaxf(x, __int_as_float(__builtin_amdgcn_ds_swizzle(__float_as_int(x), 0x081F)));
  x = fmaxf(x, __int_as_float(__builtin_amdgcn_ds_swizzle(__float_as_int(x), 0x101F)));
  x = fmaxf(x, __int_as_float(__builtin_amdgcn_ds_swizzle(__float_as_int(x), 0x201F)));
  return x;
}
__device__ __forceinline__ float redsum16(float x) {
  x += __int_as_float(__builtin_amdgcn_ds_swizzle(__float_as_int(x), 0x041F));
  x += __int_as_float(__builtin_amdgcn_ds_swizzle(__float_as_int(x), 0x081F));
  x += __int_as_float(__builtin_amdgcn_ds_swizzle(__float_as_int(x), 0x101F));
  x += __int_as_float(__builtin_amdgcn_ds_swizzle(__float_as_int(x), 0x201F));
  return x;
}

// ---------------- kernel 1: convert x to bf16, transpose weights to bf16 ----------------
__global__ void prep_kernel(const float* __restrict__ x,
                            const float* __restrict__ Wq,
                            const float* __restrict__ Wkv,
                            const float* __restrict__ Wout,
                            __bf16* __restrict__ xbf,
                            __bf16* __restrict__ wqkv_t,   // [3072][1024]
                            __bf16* __restrict__ wout_t) { // [1024][1024]
  const long n_x  = (long)ROWS * DIM;
  const long n_w1 = (long)(3 * INNER) * DIM;
  const long n_w2 = (long)INNER * DIM;
  const long total = n_x + n_w1 + n_w2;
  for (long t = (long)blockIdx.x * blockDim.x + threadIdx.x; t < total;
       t += (long)gridDim.x * blockDim.x) {
    if (t < n_x) {
      xbf[t] = (__bf16)x[t];
    } else if (t < n_x + n_w1) {
      long r = t - n_x;
      int c = (int)(r / DIM), k = (int)(r % DIM);
      float v = (c < INNER) ? Wq[(long)k * INNER + c]
                            : Wkv[(long)k * (2 * INNER) + (c - INNER)];
      wqkv_t[r] = (__bf16)v;
    } else {
      long r = t - n_x - n_w1;
      int c = (int)(r / DIM), k = (int)(r % DIM);
      wout_t[r] = (__bf16)Wout[(long)k * INNER + c];
    }
  }
}

// ---------------- kernel 2/5: WMMA GEMM, C = A(ROWSxDIM) * Bt^T  (Bt stored [col][k]) ----------------
// mode 0: QKV epilogue (scatter q/k raw bf16, v -> v_bfT, raw k/v fp32 -> next_xl output)
// mode 1: plain fp32 store to outF
__global__ __launch_bounds__(128) void gemm_kernel(
    const __bf16* __restrict__ A, const __bf16* __restrict__ Bt, int mode,
    __bf16* __restrict__ q_raw, __bf16* __restrict__ k_raw,
    __bf16* __restrict__ v_bfT, float* __restrict__ xl_out, float* __restrict__ outF) {
  const int lane = threadIdx.x & 31, wave = threadIdx.x >> 5;
  const int h16 = lane >> 4, l16 = lane & 15;
  const int m_base = blockIdx.x * 128 + (wave >> 1) * 64;
  const int c_base = blockIdx.y * 64 + (wave & 1) * 32;

  v8f vz = {};
  v8f acc[4][2];
#pragma unroll
  for (int i = 0; i < 4; ++i)
#pragma unroll
    for (int j = 0; j < 2; ++j) acc[i][j] = vz;

  for (int kk = 0; kk < DIM; kk += 32) {
    FragB16 a[4], b[2];
#pragma unroll
    for (int mt = 0; mt < 4; ++mt) {
      const __bf16* p = A + (long)(m_base + mt * 16 + l16) * DIM + kk + 8 * h16;
      a[mt].u[0] = *(const u32x4*)(p);
      a[mt].u[1] = *(const u32x4*)(p + 16);
      if (kk + 128 < DIM) __builtin_prefetch(p + 128, 0, 3);
    }
#pragma unroll
    for (int nt = 0; nt < 2; ++nt) {
      const __bf16* p = Bt + (long)(c_base + nt * 16 + l16) * DIM + kk + 8 * h16;
      b[nt].u[0] = *(const u32x4*)(p);
      b[nt].u[1] = *(const u32x4*)(p + 16);
      if (kk + 128 < DIM) __builtin_prefetch(p + 128, 0, 3);
    }
#pragma unroll
    for (int mt = 0; mt < 4; ++mt)
#pragma unroll
      for (int nt = 0; nt < 2; ++nt)
        acc[mt][nt] = WMMA_BF16(a[mt].v, b[nt].v, acc[mt][nt]);
  }

  // epilogue: C element (vgpr v, lane): m_local = v + 8*h16, n_local = l16
#pragma unroll
  for (int mt = 0; mt < 4; ++mt) {
#pragma unroll
    for (int nt = 0; nt < 2; ++nt) {
      const int c = c_base + nt * 16 + l16;
#pragma unroll
      for (int v = 0; v < 8; ++v) {
        const float f = acc[mt][nt][v];
        const int r = m_base + mt * 16 + 8 * h16 + v;  // global row (b*N + n)
        if (mode == 1) {
          outF[(long)r * INNER + c] = f;
          continue;
        }
        const int bb = r >> 11, n = r & (N - 1);
        if (c < INNER) {  // Q
          const int h = c >> 6, d = c & 63;
          q_raw[((long)(bb * H + h) * N + n) * DH + d] = (__bf16)f;
        } else if (c < 2 * INNER) {  // K
          const int cc = c - INNER, h = cc >> 6, d = cc & 63;
          k_raw[((long)(bb * H + h) * N + n) * DH + d] = (__bf16)f;
          xl_out[(((long)bb * H + h) * 2049 + (1 + n)) * DH + d] = f;  // s=0 slab
        } else {  // V
          const int cc = c - 2 * INNER, h = cc >> 6, d = cc & 63;
          v_bfT[((long)(bb * H + h) * DH + d) * KVP + (Mm + 1 + n)] = (__bf16)f;
          xl_out[(((long)(Bsz + bb) * H + h) * 2049 + (1 + n)) * DH + d] = f;  // s=1 slab
        }
      }
    }
  }
}

// ---------------- kernel 3: rotary on K, assemble k_bf / v_bfT xl+null parts, next_xl null rows ----------------
__global__ void fill_kernel(const float* __restrict__ rotary_k,
                            const float* __restrict__ xl_mem,
                            const float* __restrict__ null_kv,
                            const __bf16* __restrict__ k_raw,
                            __bf16* __restrict__ k_bf,   // [B*H][KVP][DH]
                            __bf16* __restrict__ v_bfT,  // [B*H][DH][KVP]
                            float* __restrict__ xl_out) {
  const long nKp = (long)Bsz * H * N * DH;   // projected K (rotary, j = 513+n)
  const long nKx = (long)Bsz * H * Mm * DH;  // xl K (rotary, j < 512)
  const long nKn = (long)Bsz * H * DH;       // null K (j = 512)
  const long nVx = nKx;                      // xl V -> transposed
  const long nVn = nKn;                      // null V
  const long nXn = 2L * Bsz * H * DH;        // next_xl null rows
  const long total = nKp + nKx + nKn + nVx + nVn + nXn;
  for (long t = (long)blockIdx.x * blockDim.x + threadIdx.x; t < total;
       t += (long)gridDim.x * blockDim.x) {
    long u = t;
    if (u < nKp) {
      const long bh = u / ((long)N * DH);
      const long rem = u % ((long)N * DH);
      const int n = (int)(rem / DH), d = (int)(rem % DH);
      const int j = Mm + 1 + n;
      const float tv = (float)k_raw[u];
      const float tp = (float)k_raw[u - d + (d ^ 32)];
      const float pr = (d < 32) ? -tp : tp;
      const float ang = rotary_k[(long)j * DH + d];
      k_bf[((long)bh * KVP + j) * DH + d] = (__bf16)(tv * __cosf(ang) + pr * __sinf(ang));
      continue;
    }
    u -= nKp;
    if (u < nKx) {
      const long bh = u / ((long)Mm * DH);
      const long rem = u % ((long)Mm * DH);
      const int j = (int)(rem / DH), d = (int)(rem % DH);
      const float tv = xl_mem[u];                  // s=0 slab, (b,h,j,d)
      const float tp = xl_mem[u - d + (d ^ 32)];
      const float pr = (d < 32) ? -tp : tp;
      const float ang = rotary_k[(long)j * DH + d];
      k_bf[((long)bh * KVP + j) * DH + d] = (__bf16)(tv * __cosf(ang) + pr * __sinf(ang));
      continue;
    }
    u -= nKx;
    if (u < nKn) {
      const long bh = u / DH;
      const int d = (int)(u % DH);
      const long h = bh % H;
      const float tv = null_kv[h * DH + d];
      const float tp = null_kv[h * DH + (d ^ 32)];
      const float pr = (d < 32) ? -tp : tp;
      const float ang = rotary_k[(long)Mm * DH + d];  // j = 512
      k_bf[((long)bh * KVP + Mm) * DH + d] = (__bf16)(tv * __cosf(ang) + pr * __sinf(ang));
      continue;
    }
    u -= nKn;
    if (u < nVx) {
      const long bh = u / ((long)Mm * DH);
      const long rem = u % ((long)Mm * DH);
      const int j = (int)(rem / DH), d = (int)(rem % DH);
      v_bfT[((long)bh * DH + d) * KVP + j] = (__bf16)xl_mem[(long)Bsz * H * Mm * DH + u];
      continue;
    }
    u -= nVx;
    if (u < nVn) {
      const long bh = u / DH;
      const int d = (int)(u % DH);
      const long h = bh % H;
      v_bfT[((long)bh * DH + d) * KVP + Mm] = (__bf16)null_kv[(long)H * DH + h * DH + d];
      continue;
    }
    u -= nVn;
    {  // next_xl null rows (s,b,h,d) at j=0
      const long s = u / ((long)Bsz * H * DH);
      const long rem = u % ((long)Bsz * H * DH);
      const long b = rem / (H * DH);
      const long hd = rem % (H * DH);
      const long h = hd / DH;
      const int d = (int)(hd % DH);
      xl_out[(((s * Bsz + b) * H + h) * 2049 + 0) * DH + d] = null_kv[s * H * DH + h * DH + d];
    }
  }
}

// ---------------- kernel 4: flash attention (one wave = one 16-row query tile) ----------------
// 64-key chunks: 16 WMMA per chunk vs one set of per-row xor-reductions.
__global__ __launch_bounds__(128) void attn_kernel(
    const __bf16* __restrict__ q_raw,  const float* __restrict__ rotary_q,
    const __bf16* __restrict__ k_bf,   const __bf16* __restrict__ v_bfT,
    __bf16* __restrict__ O_bf) {
  __shared__ __bf16 plds[4][16 * 72];  // per-wave 16x64 P tile, rows padded to 72 bf16
  const int lane = threadIdx.x & 31, wave = threadIdx.x >> 5;
  const int h16 = lane >> 4, l16 = lane & 15;
  const int bh = blockIdx.x >> 5;   // 32 q-blocks per (b,h)
  const int qblk = blockIdx.x & 31;
  const int b = bh >> 4, h = bh & 15;
  const int i0 = qblk * 64 + wave * 16;

  const __bf16* kbase = k_bf + (long)bh * KVP * DH;
  const __bf16* vbase = v_bfT + (long)bh * DH * KVP;

  // Q A-fragments with rotary applied on-the-fly (partner d^32 is the same slot of the other frag)
  FragB16 qa[2];
  {
    const __bf16* qr = q_raw + ((long)bh * N + i0 + l16) * DH;
    FragB16 raw[2];
#pragma unroll
    for (int f = 0; f < 2; ++f) {
      raw[f].u[0] = *(const u32x4*)(qr + 32 * f + 8 * h16);
      raw[f].u[1] = *(const u32x4*)(qr + 32 * f + 8 * h16 + 16);
    }
    const float* rq = rotary_q + (long)(i0 + l16) * DH;
    const int k0 = 8 * h16;
#pragma unroll
    for (int f = 0; f < 2; ++f)
#pragma unroll
      for (int s = 0; s < 16; ++s) {
        const int K = (s < 8) ? (k0 + s) : (k0 + s + 8);
        const int d = 32 * f + K;
        const float t0 = (float)raw[f].v[s];
        const float tp = (float)raw[1 - f].v[s];
        const float pr = (f == 0) ? -tp : tp;
        const float ang = rq[d];
        qa[f].v[s] = (__bf16)(t0 * __cosf(ang) + pr * __sinf(ang));
      }
  }

  v8f vz = {};
  v8f oacc[4];
#pragma unroll
  for (int i = 0; i < 4; ++i) oacc[i] = vz;
  float rmax[8], rsum[8];
#pragma unroll
  for (int v = 0; v < 8; ++v) { rmax[v] = -3.0e38f; rsum[v] = 0.f; }

  int kv_end = i0 + 15 + (KV - N) + 1;  // exclusive; mask handles the diagonal tail
  if (kv_end > KV) kv_end = KV;

  for (int j0 = 0; j0 < kv_end; j0 += 64) {
    // ---- QK^T: 4 key tiles x 2 d-halves ----
    v8f s[4];
#pragma unroll
    for (int t = 0; t < 4; ++t) {
      v8f c = vz;
      const __bf16* kr = kbase + (long)(j0 + 16 * t + l16) * DH;
      __builtin_prefetch(kr + 64 * DH, 0, 3);
#pragma unroll
      for (int f = 0; f < 2; ++f) {
        FragB16 kb;
        kb.u[0] = *(const u32x4*)(kr + 32 * f + 8 * h16);
        kb.u[1] = *(const u32x4*)(kr + 32 * f + 8 * h16 + 16);
        c = WMMA_BF16(qa[f].v, kb.v, c);
      }
      s[t] = c;
    }
    // ---- scale + bottom-right causal mask + online softmax ----
    const bool needMask = (j0 + 63 > i0 + (KV - N));
    float sold[8];
#pragma unroll
    for (int v = 0; v < 8; ++v) {
      const int i_row = i0 + v + 8 * h16;
      float mx;
      if (needMask) {
#pragma unroll
        for (int t = 0; t < 4; ++t) {
          const int j = j0 + 16 * t + l16;
          float sv = s[t][v] * 0.125f;  // DH^-0.5
          s[t][v] = (j <= i_row + (KV - N)) ? sv : -1.0e30f;
        }
      } else {
#pragma unroll
        for (int t = 0; t < 4; ++t) s[t][v] = s[t][v] * 0.125f;
      }
      mx = fmaxf(fmaxf(s[0][v], s[1][v]), fmaxf(s[2][v], s[3][v]));
      const float cmax = redmax16(mx);
      const float nm = fmaxf(rmax[v], cmax);
      sold[v] = __expf(rmax[v] - nm);
      rmax[v] = nm;
      float ls = 0.f;
#pragma unroll
      for (int t = 0; t < 4; ++t) {
        const float p = __expf(s[t][v] - nm);
        s[t][v] = p;
        ls += p;
      }
      rsum[v] = rsum[v] * sold[v] + redsum16(ls);
    }
#pragma unroll
    for (int dt = 0; dt < 4; ++dt)
#pragma unroll
      for (int v = 0; v < 8; ++v) oacc[dt][v] *= sold[v];

    // ---- P: C-layout -> A-layout via per-wave LDS bounce (16x64 tile) ----
    __bf16* pw = &plds[wave][0];
#pragma unroll
    for (int t = 0; t < 4; ++t)
#pragma unroll
      for (int v = 0; v < 8; ++v)
        pw[(v + 8 * h16) * 72 + 16 * t + l16] = (__bf16)s[t][v];
    wait_dscnt0();
    FragB16 pa[2];
    {
      const __bf16* pr = pw + (long)l16 * 72;
#pragma unroll
      for (int f = 0; f < 2; ++f) {
        pa[f].u[0] = *(const u32x4*)(pr + 32 * f + 8 * h16);
        pa[f].u[1] = *(const u32x4*)(pr + 32 * f + 8 * h16 + 16);
      }
    }
    // ---- PV: 4 d-tiles x 2 key halves ----
#pragma unroll
    for (int dt = 0; dt < 4; ++dt) {
      const __bf16* vr = vbase + (long)(16 * dt + l16) * KVP + j0 + 8 * h16;
      __builtin_prefetch(vr + 64, 0, 3);
#pragma unroll
      for (int f = 0; f < 2; ++f) {
        FragB16 vb;
        vb.u[0] = *(const u32x4*)(vr + 32 * f);
        vb.u[1] = *(const u32x4*)(vr + 32 * f + 16);
        oacc[dt] = WMMA_BF16(pa[f].v, vb.v, oacc[dt]);
      }
    }
  }

  // ---- finalize: O[b, i, h*64+d] in bf16 for the output GEMM ----
#pragma unroll
  for (int dt = 0; dt < 4; ++dt)
#pragma unroll
    for (int v = 0; v < 8; ++v) {
      const int i_row = i0 + v + 8 * h16;
      const int d = 16 * dt + l16;
      O_bf[((long)b * N + i_row) * INNER + h * DH + d] = (__bf16)(oacc[dt][v] / rsum[v]);
    }
}

// ---------------- host launcher ----------------
extern "C" void kernel_launch(void* const* d_in, const int* in_sizes, int n_in,
                              void* d_out, int out_size, void* d_ws, size_t ws_size,
                              hipStream_t stream) {
  const float* x        = (const float*)d_in[0];
  const float* rotary_q = (const float*)d_in[1];
  const float* rotary_k = (const float*)d_in[2];
  const float* xl_mem   = (const float*)d_in[3];
  const float* Wq       = (const float*)d_in[4];
  const float* Wkv      = (const float*)d_in[5];
  const float* Wout     = (const float*)d_in[6];
  const float* null_kv  = (const float*)d_in[7];
  float* outF   = (float*)d_out;
  float* xl_out = outF + (long)Bsz * N * INNER;  // next_xl_memories region

  char* ws = (char*)d_ws;
  size_t off = 0;
  auto alloc = [&](size_t bytes) -> void* {
    void* p = ws + off;
    off += (bytes + 255) & ~(size_t)255;
    return p;
  };
  __bf16* xbf    = (__bf16*)alloc((size_t)ROWS * DIM * 2);             // 8.0 MB
  __bf16* wqkv_t = (__bf16*)alloc((size_t)3 * INNER * DIM * 2);        // 6.0 MB
  __bf16* wout_t = (__bf16*)alloc((size_t)INNER * DIM * 2);            // 2.0 MB
  __bf16* q_raw  = (__bf16*)alloc((size_t)Bsz * H * N * DH * 2);       // 8.0 MB
  __bf16* k_raw  = (__bf16*)alloc((size_t)Bsz * H * N * DH * 2);       // 8.0 MB
  __bf16* k_bf   = (__bf16*)alloc((size_t)Bsz * H * KVP * DH * 2);     // 10.25 MB
  __bf16* v_bfT  = (__bf16*)alloc((size_t)Bsz * H * DH * KVP * 2);     // 10.25 MB
  __bf16* O_bf   = (__bf16*)alloc((size_t)ROWS * INNER * 2);           // 8.0 MB
  (void)in_sizes; (void)n_in; (void)out_size; (void)ws_size;

  prep_kernel<<<2048, 256, 0, stream>>>(x, Wq, Wkv, Wout, xbf, wqkv_t, wout_t);
  gemm_kernel<<<dim3(ROWS / 128, (3 * INNER) / 64), 128, 0, stream>>>(
      xbf, wqkv_t, 0, q_raw, k_raw, v_bfT, xl_out, nullptr);
  fill_kernel<<<4096, 256, 0, stream>>>(rotary_k, xl_mem, null_kv, k_raw, k_bf, v_bfT, xl_out);
  attn_kernel<<<Bsz * H * (N / 64), 128, 0, stream>>>(q_raw, rotary_q, k_bf, v_bfT, O_bf);
  gemm_kernel<<<dim3(ROWS / 128, INNER / 64), 128, 0, stream>>>(
      O_bf, wout_t, 1, nullptr, nullptr, nullptr, nullptr, outF);
}
